// DMPNNConvAtomMessage_7619271983742
// MI455X (gfx1250) — matrix operand order repaired and checked by reading, not compiled, not run on hardware
//
#include <hip/hip_runtime.h>
#include <hip/hip_bf16.h>

#define A_DIM 256
#define B_DIM 128
#define D_DIM 512
#define DEPTH 6

typedef __attribute__((ext_vector_type(16))) __bf16 v16bf;
typedef __attribute__((ext_vector_type(8)))  __bf16 v8bf;
typedef __attribute__((ext_vector_type(4)))  __bf16 v4bf;
typedef __attribute__((ext_vector_type(8)))  float  v8f;
typedef __attribute__((ext_vector_type(4)))  float  f32x4;

union BF16Frag { v16bf v; v8bf h[2]; };

// ---- CDNA5 async vmem->LDS (ASYNCcnt-tracked); lane provides LDS offset ----
__device__ __forceinline__ void async_b128_to_lds(const __bf16* gptr, __bf16* lptr) {
  unsigned int ldsoff = (unsigned int)(unsigned long long)lptr; // low 32b of generic = AS3 offset
  asm volatile("global_load_async_to_lds_b128 %0, %1, off"
               :: "v"(ldsoff), "v"(gptr) : "memory");
}
__device__ __forceinline__ void wait_asynccnt0() {
  asm volatile("s_wait_asynccnt 0" ::: "memory");
}

// ---------------------------------------------------------------------------
// Weight prep: Wt[n*K + k] = bf16(W[k*512 + n])   (transpose + fp32->bf16)
// ---------------------------------------------------------------------------
__global__ void wt_transpose_kernel(const float* __restrict__ W,
                                    __bf16* __restrict__ Wt, int K) {
  int i = blockIdx.x * blockDim.x + threadIdx.x;
  if (i >= K * D_DIM) return;
  int n = i / K;
  int k = i - n * K;
  Wt[i] = (__bf16)W[k * D_DIM + n];
}

__global__ void fill_zero4_kernel(float* __restrict__ p, long long nvec) {
  long long i = blockIdx.x * (long long)blockDim.x + threadIdx.x;
  if (i < nvec) *(f32x4*)(p + i * 4) = (f32x4){0.f, 0.f, 0.f, 0.f};
}

// zero columns [0, NC) of a [rows x ld] fp32 matrix (vectorized by 4)
template <int NC>
__global__ void zero_cols_kernel(float* __restrict__ p, int ld, long long totalVec) {
  long long i = blockIdx.x * (long long)blockDim.x + threadIdx.x;
  if (i >= totalVec) return;
  constexpr int NV = NC / 4;
  long long row = i / NV;
  int c = (int)(i - row * NV) * 4;
  *(f32x4*)(p + row * ld + c) = (f32x4){0.f, 0.f, 0.f, 0.f};
}

// hbuf[row][0..255] = atom[row], hbuf[row][256..767] = 0  (vectorized by 4)
__global__ void init_hbuf_kernel(const float* __restrict__ atom,
                                 float* __restrict__ hbuf, long long totalVec) {
  long long i = blockIdx.x * (long long)blockDim.x + threadIdx.x;
  if (i >= totalVec) return;
  long long row = i / 192;                // 768/4
  int c4 = (int)(i - row * 192);
  f32x4 v = (f32x4){0.f, 0.f, 0.f, 0.f};
  if (c4 < 64) v = *(const f32x4*)(atom + row * A_DIM + c4 * 4);
  *(f32x4*)(hbuf + row * 768 + c4 * 4) = v;
}

// ---------------------------------------------------------------------------
// Segment-sum scatter: out[dst[e]][col_off+c..c+3] += srcbuf[(src?src[e]:e)][c..]
// 4 channels/thread: one b128 source load + one index load per 4 atomics.
// ---------------------------------------------------------------------------
template <int NC>
__global__ void scatter_add_kernel(const float* __restrict__ srcbuf, int src_ld,
                                   const int* __restrict__ src_idx,  // nullable
                                   const int* __restrict__ dst_idx,
                                   float* __restrict__ out, int out_ld,
                                   int col_off, long long totalVec) {
  long long i = blockIdx.x * (long long)blockDim.x + threadIdx.x;
  if (i >= totalVec) return;
  constexpr int NV = NC / 4;
  long long e = i / NV;
  int c = (int)(i - e * NV) * 4;
  int r = src_idx ? src_idx[e] : (int)e;
  f32x4 v = *(const f32x4*)(srcbuf + (long long)r * src_ld + c);
  float* o = out + (long long)dst_idx[e] * out_ld + col_off + c;
  atomicAdd(o + 0, v.x);
  atomicAdd(o + 1, v.y);
  atomicAdd(o + 2, v.z);
  atomicAdd(o + 3, v.w);
}

// ---------------------------------------------------------------------------
// Tiled bf16 WMMA GEMM:  Out = relu(Cin_or_bias + A @ W)
// Double-buffered LDS; A staged via regs (fp32->bf16 convertvector);
// B staged via global_load_async_to_lds_b128. Rows clamped (no exec branches);
// only the epilogue store is guarded.
// Block: 256 threads (8 waves). Tile: BM=128 (16 rows/wave), BN=64, BK=32.
// ---------------------------------------------------------------------------
#define BM 128
#define BN 64
#define BK 32
#define LPAD 40   // LDS row stride in bf16 (80B: 16B-aligned, bank-spread)

__global__ __launch_bounds__(256)
void gemm_wmma_kernel(const float* __restrict__ Amat, int lda, int K, int Nrows,
                      const __bf16* __restrict__ Wt,
                      const float* __restrict__ Cin,
                      const float* __restrict__ bias,
                      float* __restrict__ OutRelu,
                      float* __restrict__ OutPre) {
  __shared__ __attribute__((aligned(16))) __bf16 lA[2][BM * LPAD];
  __shared__ __attribute__((aligned(16))) __bf16 lB[2][BN * LPAD];

  const int t    = threadIdx.x;
  const int wave = t >> 5;
  const int lane = t & 31;
  const int m0   = blockIdx.x * BM;
  const int n0   = blockIdx.y * BN;
  const int fm   = lane & 15;   // row (A) / col (B,C) within 16
  const int hi   = lane >> 4;   // half-wave select
  const int kbA  = hi * 8;      // A frag K base per ISA 16-bit A layout
  const int kbB  = hi * 8;      // B frag K base, same K order as A

  // ---- per-thread staging coordinates (invariant over K loop) ----
  int ar[4], ac4[4];
  const float* aptr[4];
#pragma unroll
  for (int j = 0; j < 4; ++j) {
    int idx = j * 256 + t;      // 1024 float4s per 128x32 tile
    ar[j]  = idx >> 3;          // 8 float4 per row
    ac4[j] = idx & 7;
    int row = m0 + ar[j];
    if (row > Nrows - 1) row = Nrows - 1;   // clamp: OOB rows never stored
    aptr[j] = Amat + (long long)row * lda + ac4[j] * 4;
  }
  const int bn  = t >> 2;       // 0..63
  const int bkk = (t & 3) * 8;  // 0,8,16,24
  const __bf16* bptr = Wt + (long long)(n0 + bn) * K + bkk;

  // ---- accumulator init: 0, bias broadcast, or residual Cin (clamped) ----
  v8f acc[4];
#pragma unroll
  for (int nt = 0; nt < 4; ++nt) {
#pragma unroll
    for (int v = 0; v < 8; ++v) {
      int row = m0 + wave * 16 + hi * 8 + v;
      if (row > Nrows - 1) row = Nrows - 1;
      int col = n0 + nt * 16 + fm;
      float cv = 0.0f;
      if (bias) cv = bias[col];
      if (Cin)  cv = Cin[(long long)row * D_DIM + col];
      acc[nt][v] = cv;
    }
  }

  const int nk = K / BK;
  // ---- prologue: stage tile 0 into buffer 0 ----
  {
    f32x4 f[4];
#pragma unroll
    for (int j = 0; j < 4; ++j) f[j] = *(const f32x4*)(aptr[j]);
    async_b128_to_lds(bptr, &lB[0][bn * LPAD + bkk]);
#pragma unroll
    for (int j = 0; j < 4; ++j)
      *(v4bf*)(&lA[0][ar[j] * LPAD + ac4[j] * 4]) = __builtin_convertvector(f[j], v4bf);
    wait_asynccnt0();
  }
  __syncthreads();

  for (int kb = 0; kb < nk; ++kb) {
    const int cur  = kb & 1;
    const int nxt  = cur ^ 1;
    const bool more = (kb + 1) < nk;

    // issue next tile's global traffic before compute (latency hidden by WMMA)
    f32x4 f[4];
    if (more) {
      const long long koff = (long long)(kb + 1) * BK;
#pragma unroll
      for (int j = 0; j < 4; ++j) f[j] = *(const f32x4*)(aptr[j] + koff);
      async_b128_to_lds(bptr + koff, &lB[nxt][bn * LPAD + bkk]);
    }

    // ---- WMMA on current buffers ----
    BF16Frag af;
    const __bf16* pA = &lA[cur][(wave * 16 + fm) * LPAD + kbA];
    af.h[0] = *(const v8bf*)(pA);        // K = kbA .. kbA+7
    af.h[1] = *(const v8bf*)(pA + 16);   // K = kbA+16 .. kbA+23
#pragma unroll
    for (int nt = 0; nt < 4; ++nt) {
      BF16Frag bfr;
      const __bf16* pB = &lB[cur][(nt * 16 + fm) * LPAD + kbB];
      bfr.h[0] = *(const v8bf*)(pB);     // K = kbB .. +7
      bfr.h[1] = *(const v8bf*)(pB + 16); // K = kbB+16 .. +23
      acc[nt] = __builtin_amdgcn_wmma_f32_16x16x32_bf16(
          false, af.v, false, bfr.v, (short)0, acc[nt], false, false);
    }

    // commit next tile to LDS after compute
    if (more) {
#pragma unroll
      for (int j = 0; j < 4; ++j)
        *(v4bf*)(&lA[nxt][ar[j] * LPAD + ac4[j] * 4]) = __builtin_convertvector(f[j], v4bf);
      wait_asynccnt0();
    }
    __syncthreads();
  }

  // ---- epilogue: optional pre-ReLU store + ReLU store (guarded) ----
#pragma unroll
  for (int nt = 0; nt < 4; ++nt) {
#pragma unroll
    for (int v = 0; v < 8; ++v) {
      int row = m0 + wave * 16 + hi * 8 + v;
      if (row < Nrows) {
        int col = n0 + nt * 16 + fm;
        float val = acc[nt][v];
        long long o = (long long)row * D_DIM + col;
        if (OutPre) OutPre[o] = val;
        OutRelu[o] = val > 0.0f ? val : 0.0f;
      }
    }
  }
}

// ---------------------------------------------------------------------------
// Orchestration
// ---------------------------------------------------------------------------
extern "C" void kernel_launch(void* const* d_in, const int* in_sizes, int n_in,
                              void* d_out, int out_size, void* d_ws, size_t ws_size,
                              hipStream_t stream) {
  const float* atom = (const float*)d_in[0];
  const float* bond = (const float*)d_in[1];
  const int*   src  = (const int*)d_in[2];
  const int*   dst  = (const int*)d_in[3];
  const float* W_i  = (const float*)d_in[4];
  const float* W_h  = (const float*)d_in[5];
  const float* W_o  = (const float*)d_in[6];
  const float* b_o  = (const float*)d_in[7];

  const int N = in_sizes[0] / A_DIM;
  const int E = in_sizes[2];
  const int Kh = B_DIM + D_DIM;   // 640
  const int Ko = A_DIM + D_DIM;   // 768

  char* ws = (char*)d_ws;
  size_t off = 0;
  auto alloc = [&](size_t bytes) -> void* {
    void* p = ws + off;
    off = (off + bytes + 255) & ~(size_t)255;
    return p;
  };
  __bf16* WiT = (__bf16*)alloc((size_t)D_DIM * A_DIM * 2);
  __bf16* WhT = (__bf16*)alloc((size_t)D_DIM * Kh * 2);
  __bf16* WoT = (__bf16*)alloc((size_t)D_DIM * Ko * 2);
  float*  inp = (float*)alloc((size_t)N * D_DIM * 4);   // pre-ReLU residual
  float*  msg = (float*)alloc((size_t)N * D_DIM * 4);   // message (post-ReLU)
  float*  big = (float*)alloc((size_t)N * Ko * 4);      // agg [Nx640] / h [Nx768]
  float*  agg  = big;
  float*  hbuf = big;

  const int thr = 256;
  // 1) weight prep (transpose + bf16)
  wt_transpose_kernel<<<(D_DIM * A_DIM + thr - 1) / thr, thr, 0, stream>>>(W_i, WiT, A_DIM);
  wt_transpose_kernel<<<(D_DIM * Kh + thr - 1) / thr, thr, 0, stream>>>(W_h, WhT, Kh);
  wt_transpose_kernel<<<(D_DIM * Ko + thr - 1) / thr, thr, 0, stream>>>(W_o, WoT, Ko);

  // 2) agg = 0; scatter bond into agg[:, 512:640] (constant across depth)
  long long aggVec = (long long)N * Kh / 4;
  fill_zero4_kernel<<<(unsigned)((aggVec + thr - 1) / thr), thr, 0, stream>>>(agg, aggVec);
  long long tbv = (long long)E * B_DIM / 4;
  scatter_add_kernel<B_DIM><<<(unsigned)((tbv + thr - 1) / thr), thr, 0, stream>>>(
      bond, B_DIM, nullptr, dst, agg, Kh, D_DIM, tbv);

  // 3) inp = atom @ W_i ; msg = relu(inp)
  dim3 gg((N + BM - 1) / BM, D_DIM / BN);
  gemm_wmma_kernel<<<gg, thr, 0, stream>>>(atom, A_DIM, A_DIM, N, WiT,
                                           nullptr, nullptr, msg, inp);

  // 4) depth loop
  long long thv = (long long)E * D_DIM / 4;
  long long znv = (long long)N * D_DIM / 4;
  for (int d = 0; d < DEPTH - 1; ++d) {
    zero_cols_kernel<D_DIM><<<(unsigned)((znv + thr - 1) / thr), thr, 0, stream>>>(agg, Kh, znv);
    scatter_add_kernel<D_DIM><<<(unsigned)((thv + thr - 1) / thr), thr, 0, stream>>>(
        msg, D_DIM, src, dst, agg, Kh, 0, thv);
    gemm_wmma_kernel<<<gg, thr, 0, stream>>>(agg, Kh, Kh, N, WhT,
                                             inp, nullptr, msg, nullptr);
  }

  // 5) h = [atom | segment_sum(msg)] ; out = relu(h @ W_o + b_o)
  long long hnv = (long long)N * Ko / 4;
  init_hbuf_kernel<<<(unsigned)((hnv + thr - 1) / thr), thr, 0, stream>>>(atom, hbuf, hnv);
  scatter_add_kernel<D_DIM><<<(unsigned)((thv + thr - 1) / thr), thr, 0, stream>>>(
      msg, D_DIM, src, dst, hbuf, Ko, A_DIM, thv);
  gemm_wmma_kernel<<<gg, thr, 0, stream>>>(hbuf, Ko, Ko, N, WoT,
                                           nullptr, b_o, (float*)d_out, nullptr);
}
